// LoRALinear_60129542144689
// MI455X (gfx1250) — matrix operand order repaired and checked by reading, not compiled, not run on hardware
//
#include <hip/hip_runtime.h>
#include <hip/hip_bf16.h>

// ---------------- problem constants (fixed by the reference) ----------------
#define IN_F   4096          // K
#define OUT_F  4096          // N
#define M_TOT  8192          // 4 * 2048
#define RANK   8
#define LORA_SCALE 4.0f      // ALPHA / RANK = 32 / 8

// ---------------- tiling ----------------
#define BM 128
#define BN 128
#define BK 64                // bf16 elements of K staged per LDS buffer (2 WMMA K-steps)
#define LSTR 72              // bf16 elems per LDS row (64 data + 8 pad) -> 144B stride, 16B aligned

typedef __bf16  bf16;
typedef __bf16  v8bf  __attribute__((ext_vector_type(8)));
typedef __bf16  v16bf __attribute__((ext_vector_type(16)));
typedef float   v8f   __attribute__((ext_vector_type(8)));

// async-copy pointer types (match builtin signature: int4-vector ptrs, AS1/AS3)
typedef int v4i __attribute__((vector_size(16)));
typedef __attribute__((address_space(1))) v4i v4i_as1;
typedef __attribute__((address_space(3))) v4i v4i_as3;

#if __has_builtin(__builtin_amdgcn_global_load_async_to_lds_b128) && \
    __has_builtin(__builtin_amdgcn_s_wait_asynccnt)
#define USE_ASYNC 1
#else
#define USE_ASYNC 0
#endif

// ============================================================================
// Kernel 0: elementwise f32 -> bf16 (RNE via v_cvt_pk), 8 elements / thread
// ============================================================================
__global__ __launch_bounds__(256) void cvt_bf16_kernel(
    const float* __restrict__ src, bf16* __restrict__ dst) {
  size_t i = ((size_t)blockIdx.x * 256 + threadIdx.x) * 8;
  float4 lo = *(const float4*)(src + i);
  float4 hi = *(const float4*)(src + i + 4);
  v8bf v = { (bf16)lo.x, (bf16)lo.y, (bf16)lo.z, (bf16)lo.w,
             (bf16)hi.x, (bf16)hi.y, (bf16)hi.z, (bf16)hi.w };
  *(v8bf*)(dst + i) = v;
}

// ============================================================================
// Kernel 1: t[m][r] = sum_k x[m][k] * b[r][k]   (8192 x 8 rank-8 pre-pass)
// One wave32 per row m; full-precision f32 inputs.
// ============================================================================
__global__ __launch_bounds__(256) void lora_t_kernel(
    const float* __restrict__ x, const float* __restrict__ bmat,
    float* __restrict__ t) {
  const int wave = threadIdx.x >> 5;
  const int lane = threadIdx.x & 31;
  const int m = blockIdx.x * 8 + wave;

  const float4* xrow = (const float4*)(x + (size_t)m * IN_F);
  float acc[RANK];
#pragma unroll
  for (int r = 0; r < RANK; ++r) acc[r] = 0.0f;

  for (int k4 = lane; k4 < IN_F / 4; k4 += 32) {
    float4 xv = xrow[k4];
#pragma unroll
    for (int r = 0; r < RANK; ++r) {
      float4 bv = ((const float4*)(bmat + (size_t)r * IN_F))[k4];
      acc[r] += xv.x * bv.x + xv.y * bv.y + xv.z * bv.z + xv.w * bv.w;
    }
  }
#pragma unroll
  for (int r = 0; r < RANK; ++r) {
    float v = acc[r];
#pragma unroll
    for (int off = 16; off > 0; off >>= 1) v += __shfl_xor(v, off, 32);
    if (lane == 0) t[m * RANK + r] = v;
  }
}

// ============================================================================
// Kernel 2: out = x @ W^T + bias + 4 * t @ a^T     (bf16 WMMA, f32 accumulate)
// 128x128 block tile, 8 wave32s; wave tile 32x64 = 2x4 WMMAs per K-step.
// BK=64 staged per LDS buffer via async global->LDS copies, double buffered.
// ============================================================================
__global__ __launch_bounds__(256) void lora_gemm_kernel(
    const bf16* __restrict__ xb, const bf16* __restrict__ wb,
    const float* __restrict__ amat, const float* __restrict__ bias,
    const float* __restrict__ t, float* __restrict__ out) {
  __shared__ bf16 As[2][BM * LSTR];
  __shared__ bf16 Bs[2][BN * LSTR];

  const int tid    = threadIdx.x;
  const int waveId = tid >> 5;
  const int lane   = tid & 31;
  const int lane16 = lane & 15;
  const int half   = lane >> 4;

  const int blockM = blockIdx.y * BM;
  const int blockN = blockIdx.x * BN;
  const int waveM  = (waveId & 3) * 32;    // 4 waves along M
  const int waveN  = (waveId >> 2) * 64;   // 2 waves along N

  v8f acc[2][4];
#pragma unroll
  for (int rt = 0; rt < 2; ++rt)
#pragma unroll
    for (int ct = 0; ct < 4; ++ct) acc[rt][ct] = (v8f){0,0,0,0,0,0,0,0};

  // stage copy: 128x64 bf16 per matrix = 16KB; 256 threads x 4 x 16B each
  auto issue_stage = [&](int buf, int k0) {
#pragma unroll
    for (int i = 0; i < 4; ++i) {
      int lin = tid + i * 256;             // 0..1023
      int row = lin >> 3;                  // 0..127
      int c8  = (lin & 7) << 3;            // 0,8,...,56 (bf16 elems, 16B chunks)
      const bf16* ga = xb + (size_t)(blockM + row) * IN_F + k0 + c8;
      const bf16* gb = wb + (size_t)(blockN + row) * IN_F + k0 + c8;
      bf16* la = &As[buf][row * LSTR + c8];
      bf16* lb = &Bs[buf][row * LSTR + c8];
#if USE_ASYNC
      __builtin_amdgcn_global_load_async_to_lds_b128(
          (v4i_as1*)ga, (v4i_as3*)la, 0, 0);
      __builtin_amdgcn_global_load_async_to_lds_b128(
          (v4i_as1*)gb, (v4i_as3*)lb, 0, 0);
#else
      uint4 va = *(const uint4*)ga;
      uint4 vb = *(const uint4*)gb;
      *(uint4*)la = va;
      *(uint4*)lb = vb;
#endif
    }
  };

  issue_stage(0, 0);                        // prologue

  const int KT = IN_F / BK;                 // 64 stages
  int buf = 0;
  for (int kt = 0; kt < KT; ++kt) {
#if USE_ASYNC
    __builtin_amdgcn_s_wait_asynccnt(0);    // own stage-kt copies landed in LDS
#endif
    __syncthreads();                        // everyone's data present; prev reads done
    if (kt + 1 < KT) issue_stage(buf ^ 1, (kt + 1) * BK);

#pragma unroll
    for (int ks = 0; ks < 2; ++ks) {        // two K=32 WMMA steps per stage
      const int kb = ks * 32;               // bf16 element offset within LDS row
      // A fragments: lane holds row M=lane16; contiguous K run half*16..+15
      v16bf afrag[2];
#pragma unroll
      for (int rt = 0; rt < 2; ++rt) {
        const bf16* ap = &As[buf][(waveM + rt * 16 + lane16) * LSTR + kb];
        v8bf lo = *(const v8bf*)(ap + half * 16);
        v8bf hi = *(const v8bf*)(ap + half * 16 + 8);
        afrag[rt] = __builtin_shufflevector(lo, hi,
            0,1,2,3,4,5,6,7,8,9,10,11,12,13,14,15);
      }
      // B fragments: lane holds col N=lane16; contiguous K run half*16..+15
      v16bf bfrag[4];
#pragma unroll
      for (int ct = 0; ct < 4; ++ct) {
        const bf16* bp = &Bs[buf][(waveN + ct * 16 + lane16) * LSTR + kb + half * 16];
        v8bf lo = *(const v8bf*)(bp);
        v8bf hi = *(const v8bf*)(bp + 8);
        bfrag[ct] = __builtin_shufflevector(lo, hi,
            0,1,2,3,4,5,6,7,8,9,10,11,12,13,14,15);
      }
#pragma unroll
      for (int rt = 0; rt < 2; ++rt)
#pragma unroll
        for (int ct = 0; ct < 4; ++ct)
          acc[rt][ct] = __builtin_amdgcn_wmma_f32_16x16x32_bf16(
              false, afrag[rt], false, bfrag[ct],
              (short)0, acc[rt][ct], false, false);
    }
    buf ^= 1;
  }

  // ---- fused epilogue: + bias[n] + 4 * dot(t[m][0:8], a[n][0:8]) ----
  // C layout: VGPR i, lanes 0-15 -> M=i, N=lane; lanes 16-31 -> M=8+i, N=lane-16
  float4 a0[4], a1[4];
  float  bn[4];
#pragma unroll
  for (int ct = 0; ct < 4; ++ct) {
    int n = blockN + waveN + ct * 16 + lane16;
    a0[ct] = *(const float4*)(amat + (size_t)n * RANK);
    a1[ct] = *(const float4*)(amat + (size_t)n * RANK + 4);
    bn[ct] = bias[n];
  }
#pragma unroll
  for (int rt = 0; rt < 2; ++rt) {
    int mbase = blockM + waveM + rt * 16 + half * 8;
#pragma unroll
    for (int i = 0; i < 8; ++i) {
      int m = mbase + i;
      float4 t0 = *(const float4*)(t + (size_t)m * RANK);
      float4 t1 = *(const float4*)(t + (size_t)m * RANK + 4);
#pragma unroll
      for (int ct = 0; ct < 4; ++ct) {
        float lora = t0.x * a0[ct].x + t0.y * a0[ct].y + t0.z * a0[ct].z + t0.w * a0[ct].w
                   + t1.x * a1[ct].x + t1.y * a1[ct].y + t1.z * a1[ct].z + t1.w * a1[ct].w;
        int n = blockN + waveN + ct * 16 + lane16;
        out[(size_t)m * OUT_F + n] = acc[rt][ct][i] + bn[ct] + LORA_SCALE * lora;
      }
    }
  }
}

// ============================================================================
extern "C" void kernel_launch(void* const* d_in, const int* in_sizes, int n_in,
                              void* d_out, int out_size, void* d_ws, size_t ws_size,
                              hipStream_t stream) {
  const float* x    = (const float*)d_in[0];
  const float* wgt  = (const float*)d_in[1];
  const float* amat = (const float*)d_in[2];
  const float* bmat = (const float*)d_in[3];
  const float* bias = (const float*)d_in[4];
  float* out = (float*)d_out;

  // workspace layout: [x_bf16 | w_bf16 | t]
  char* ws = (char*)d_ws;
  bf16*  xb = (bf16*)ws;                                        // 8192*4096*2 = 64 MiB
  bf16*  wb = (bf16*)(ws + (size_t)M_TOT * IN_F * 2);           // 4096*4096*2 = 32 MiB
  float* t  = (float*)(ws + (size_t)M_TOT * IN_F * 2
                          + (size_t)OUT_F * IN_F * 2);          // 256 KiB

  cvt_bf16_kernel<<<dim3((M_TOT * (size_t)IN_F) / 2048), dim3(256), 0, stream>>>(x, xb);
  cvt_bf16_kernel<<<dim3((OUT_F * (size_t)IN_F) / 2048), dim3(256), 0, stream>>>(wgt, wb);
  lora_t_kernel<<<dim3(M_TOT / 8), dim3(256), 0, stream>>>(x, bmat, t);
  lora_gemm_kernel<<<dim3(OUT_F / BN, M_TOT / BM), dim3(256), 0, stream>>>(
      xb, wb, amat, bias, t, out);
}
